// RNN_77146202571102
// MI455X (gfx1250) — compile-verified, matrix operand order checked by reading, not run on hardware
//
#include <hip/hip_runtime.h>
#include <hip/hip_bf16.h>
#include <math.h>

// ---------------------------------------------------------------------------
// RNN forward for MI455X (gfx1250), wave32 + WMMA bf16.
//   K_cvt  : one-pass fp32 -> bf16 for x, W_ih, W_fc (so GEMM loops are
//            pure b128 loads + v_wmma, no in-loop cvt)
//   K_init : zero grid-barrier counter, h0 -> bf16
//   K_proj : xt[T,H,B] = x[B*T,D] @ W_ih[H,D]^T + b_ih         (WMMA bf16)
//   K_scan : persistent, 32 WGs; W_hh column-slice in LDS (66 KB of 320 KB);
//            per step h = tanh(xt_t + h @ W_hh^T + b_hh)       (WMMA bf16)
//            xt slice in [H,B] order -> vectorized b128 reads on the
//            serial critical path; grid barrier via device-scope atomics
//   K_fc   : out[T,O] = seq63[T,H] @ W_fc[O,H]^T + b_fc        (WMMA bf16)
// ---------------------------------------------------------------------------

#define DEVINL __device__ __forceinline__

typedef __attribute__((ext_vector_type(16))) __bf16 v16bf;
typedef __attribute__((ext_vector_type(8)))  __bf16 bf16x8;
typedef __attribute__((ext_vector_type(8)))  float  v8f;
typedef __attribute__((ext_vector_type(4)))  float  f32x4;

constexpr int Bsz = 64;
constexpr int Tt  = 1024;
constexpr int Dd  = 512;
constexpr int Hh  = 1024;
constexpr int Oo  = 512;
constexpr int BT  = Bsz * Tt;       // 65536
constexpr int NWG = 32;             // persistent workgroups in scan
constexpr int LDSK = Hh + 8;        // padded K stride (bank-conflict avoidance)

// ---- fragment helpers -----------------------------------------------------
// 16-bit A-matrix 16x32 layout (ISA 7.12.2): lane L holds row M=L%16;
// per-lane element i maps to K = (i%8) + 8*(L>=16) + 16*(i/8).
// B (32x16, source stored row-major [N][K]) mirrors it with lane=N%16.

DEVINL void cpy8(const __bf16* p, v16bf& d, int base) {
  bf16x8 v = *reinterpret_cast<const bf16x8*>(p);
#pragma unroll
  for (int i = 0; i < 8; ++i) d[base + i] = v[i];
}

// rowp: pointer to this lane's row; kb = kbase + 8*(lane>=16)
DEVINL v16bf frag_bf16(const __bf16* rowp, int kb) {
  v16bf d; cpy8(rowp + kb, d, 0); cpy8(rowp + kb + 16, d, 8); return d;
}

DEVINL v8f wmma_bf16(v16bf a, v16bf b, v8f c) {
  // (neg_a, A, neg_b, B, c_mod, C, reuse_a, reuse_b)
  return __builtin_amdgcn_wmma_f32_16x16x32_bf16(false, a, false, b, (short)0,
                                                 c, false, false);
}

// ---- one-pass fp32 -> bf16 conversion (8 elems/thread, b128 in, b128 out) --
__global__ void __launch_bounds__(256) cvt_f32_bf16(
    const float* __restrict__ src, __bf16* __restrict__ dst, int n) {
  int i = (blockIdx.x * blockDim.x + threadIdx.x) * 8;
  if (i < n) {
    f32x4 a = *reinterpret_cast<const f32x4*>(src + i);
    f32x4 b = *reinterpret_cast<const f32x4*>(src + i + 4);
    bf16x8 o;
#pragma unroll
    for (int j = 0; j < 4; ++j) o[j] = (__bf16)a[j];
#pragma unroll
    for (int j = 0; j < 4; ++j) o[4 + j] = (__bf16)b[j];
    *reinterpret_cast<bf16x8*>(dst + i) = o;
  }
}

// ---- init -----------------------------------------------------------------
__global__ void __launch_bounds__(256) rnn_init(const float* __restrict__ hidden,
                                                __bf16* __restrict__ h0,
                                                unsigned* __restrict__ bar) {
  unsigned i = blockIdx.x * blockDim.x + threadIdx.x;
  if (i == 0) *bar = 0u;
  if (i < (unsigned)(Bsz * Hh)) h0[i] = (__bf16)hidden[i];
}

// ---- phase 1: input projection -------------------------------------------
// grid: (H/128, BT/128), block 256 (8 waves); wave tile 32x64 (2x4 WMMA tiles)
// 8 WMMA per K-chunk against 6 fragment loads.
__global__ void __launch_bounds__(256) rnn_proj_wmma(
    const __bf16* __restrict__ xb, const __bf16* __restrict__ wb,
    const float* __restrict__ b_ih, float* __restrict__ xt) {
  const int lane = threadIdx.x & 31, w = threadIdx.x >> 5;
  const int m0 = blockIdx.y * 128 + (w >> 1) * 32;  // 4 waves over M
  const int n0 = blockIdx.x * 128 + (w & 1) * 64;   // 2 waves over N
  const int lr = lane & 15;
  const int kh = (lane >> 4) << 3;

  const __bf16* a0p = xb + (size_t)(m0 + lr) * Dd;
  const __bf16* a1p = a0p + (size_t)16 * Dd;
  const __bf16* bp0 = wb + (size_t)(n0 + lr) * Dd;

  v8f acc[2][4] = {};
#pragma unroll 2
  for (int k = 0; k < Dd; k += 32) {
    v16bf a0 = frag_bf16(a0p, k + kh);
    v16bf a1 = frag_bf16(a1p, k + kh);
#pragma unroll
    for (int j = 0; j < 4; ++j) {
      v16bf b = frag_bf16(bp0 + (size_t)(16 * j) * Dd, k + kh);
      acc[0][j] = wmma_bf16(a0, b, acc[0][j]);
      acc[1][j] = wmma_bf16(a1, b, acc[1][j]);
    }
  }
  // C layout: element r of acc -> M = r + 8*(lane>=16), N = lane%16
#pragma unroll
  for (int i = 0; i < 2; ++i) {
#pragma unroll
    for (int j = 0; j < 4; ++j) {
      const int n = n0 + j * 16 + lr;
      const float bias = b_ih[n];
#pragma unroll
      for (int r = 0; r < 8; ++r) {
        int m = m0 + i * 16 + kh + r;        // row in [B*T]
        int b = m >> 10;                     // m / T
        int t = m & (Tt - 1);                // m % T
        xt[((size_t)t * Hh + n) * Bsz + b] = acc[i][j][r] + bias;  // [T,H,B]
      }
    }
  }
}

// ---- phase 2: sequential recurrence (persistent) --------------------------
// 32 WGs x 256 threads. WG g owns output columns [g*32, g*32+32).
// W_hh slice (32 x 1024 bf16, padded) lives in LDS for all 1024 steps.
// Each wave computes one 16x16 tile of the 64x32 slice output.
__global__ void __launch_bounds__(256) rnn_scan_wmma(
    const float* __restrict__ xt, const float* __restrict__ W_hh,
    const float* __restrict__ b_hh, __bf16* hbuf0, __bf16* hbuf1,
    __bf16* __restrict__ seq63, float* __restrict__ hT_out,
    unsigned* __restrict__ bar) {
  __shared__ __attribute__((aligned(16))) __bf16 whh[32][LDSK];

  const int n_wg = blockIdx.x * 32;

  // Preload + convert our W_hh column slice into LDS (reused 1024 times).
  for (int idx = threadIdx.x; idx < 32 * Hh; idx += 256) {
    int j = idx >> 10, k = idx & (Hh - 1);
    whh[j][k] = (__bf16)W_hh[(size_t)(n_wg + j) * Hh + k];
  }
  __syncthreads();

  const int lane = threadIdx.x & 31, w = threadIdx.x >> 5;
  const int m0 = (w >> 1) * 16;          // batch tile row
  const int nl0 = (w & 1) * 16;          // local column tile
  const int lr = lane & 15;
  const int kh = (lane >> 4) << 3;
  const int n_glob = n_wg + nl0 + lr;
  const float bias = b_hh[n_glob];
  const __bf16* brow = &whh[nl0 + lr][0];
  // this lane's 8 contiguous batch entries within a [H,B] step slice
  const size_t xoff = (size_t)n_glob * Bsz + m0 + kh;

  for (int t = 0; t < Tt; ++t) {
    const __bf16* cur = (t & 1) ? hbuf1 : hbuf0;
    __bf16* nxt = (t & 1) ? hbuf0 : hbuf1;
    const float* xp = xt + (size_t)t * (Hh * Bsz) + xoff;

    if (t + 1 < Tt)  // global_prefetch of next step's xt slice
      __builtin_prefetch(xp + (size_t)(Hh * Bsz), 0, 1);

    const __bf16* arow = cur + (size_t)(m0 + lr) * Hh;

    // 4 accumulators to expose WMMA ILP on the serial C-chain
    v8f acc0 = {}, acc1 = {}, acc2 = {}, acc3 = {};
#pragma unroll 2
    for (int k = 0; k < Hh; k += 128) {
      acc0 = wmma_bf16(frag_bf16(arow, k + kh), frag_bf16(brow, k + kh), acc0);
      acc1 = wmma_bf16(frag_bf16(arow, k + 32 + kh),
                       frag_bf16(brow, k + 32 + kh), acc1);
      acc2 = wmma_bf16(frag_bf16(arow, k + 64 + kh),
                       frag_bf16(brow, k + 64 + kh), acc2);
      acc3 = wmma_bf16(frag_bf16(arow, k + 96 + kh),
                       frag_bf16(brow, k + 96 + kh), acc3);
    }
    v8f acc = (acc0 + acc1) + (acc2 + acc3);

    // vectorized xt read: 8 contiguous floats (2 x b128) per lane
    f32x4 x0 = *reinterpret_cast<const f32x4*>(xp);
    f32x4 x1 = *reinterpret_cast<const f32x4*>(xp + 4);

#pragma unroll
    for (int r = 0; r < 8; ++r) {
      int m = m0 + kh + r;  // batch index
      float xv = (r < 4) ? x0[r] : x1[r - 4];
      float v = tanhf(acc[r] + xv + bias);
      nxt[(size_t)m * Hh + n_glob] = (__bf16)v;
      if (m == Bsz - 1) seq63[(size_t)t * Hh + n_glob] = (__bf16)v;
      if (t == Tt - 1) hT_out[(size_t)m * Hh + n_glob] = v;
    }

    // ---- grid-wide step barrier (monotonic counter, device scope) ----
    __syncthreads();
    if (threadIdx.x == 0) {
      __builtin_amdgcn_fence(__ATOMIC_RELEASE, "agent");
      __hip_atomic_fetch_add(bar, 1u, __ATOMIC_RELAXED,
                             __HIP_MEMORY_SCOPE_AGENT);
      const unsigned target = (unsigned)(t + 1) * NWG;
      while (__hip_atomic_load(bar, __ATOMIC_RELAXED,
                               __HIP_MEMORY_SCOPE_AGENT) < target) {
        __builtin_amdgcn_s_sleep(2);
      }
    }
    __syncthreads();
    __builtin_amdgcn_fence(__ATOMIC_ACQUIRE, "agent");
  }
}

// ---- phase 3: final FC ----------------------------------------------------
// out[T,O] = seq63[T,H] @ W_fc[O,H]^T + b_fc. grid (O/128, T/128), block 256.
__global__ void __launch_bounds__(256) rnn_fc_wmma(
    const __bf16* __restrict__ seq, const __bf16* __restrict__ wfc,
    const float* __restrict__ b_fc, float* __restrict__ out) {
  const int lane = threadIdx.x & 31, w = threadIdx.x >> 5;
  const int m0 = blockIdx.y * 128 + (w >> 1) * 32;
  const int n0 = blockIdx.x * 128 + (w & 1) * 64;
  const int lr = lane & 15;
  const int kh = (lane >> 4) << 3;

  const __bf16* a0p = seq + (size_t)(m0 + lr) * Hh;
  const __bf16* a1p = a0p + (size_t)16 * Hh;
  const __bf16* bp0 = wfc + (size_t)(n0 + lr) * Hh;

  v8f acc[2][4] = {};
#pragma unroll 2
  for (int k = 0; k < Hh; k += 32) {
    v16bf a0 = frag_bf16(a0p, k + kh);
    v16bf a1 = frag_bf16(a1p, k + kh);
#pragma unroll
    for (int j = 0; j < 4; ++j) {
      v16bf b = frag_bf16(bp0 + (size_t)(16 * j) * Hh, k + kh);
      acc[0][j] = wmma_bf16(a0, b, acc[0][j]);
      acc[1][j] = wmma_bf16(a1, b, acc[1][j]);
    }
  }
#pragma unroll
  for (int i = 0; i < 2; ++i) {
#pragma unroll
    for (int j = 0; j < 4; ++j) {
      const int n = n0 + j * 16 + lr;
      const float bias = b_fc[n];
#pragma unroll
      for (int r = 0; r < 8; ++r) {
        int m = m0 + i * 16 + kh + r;
        out[(size_t)m * Oo + n] = acc[i][j][r] + bias;
      }
    }
  }
}

// ---- host-side launcher ---------------------------------------------------
extern "C" void kernel_launch(void* const* d_in, const int* in_sizes, int n_in,
                              void* d_out, int out_size, void* d_ws,
                              size_t ws_size, hipStream_t stream) {
  (void)in_sizes; (void)n_in; (void)out_size; (void)ws_size;

  const float* x      = (const float*)d_in[0];  // [B,T,D]
  const float* hidden = (const float*)d_in[1];  // [1,B,H]
  const float* W_ih   = (const float*)d_in[2];  // [H,D]
  const float* W_hh   = (const float*)d_in[3];  // [H,H]
  const float* b_ih   = (const float*)d_in[4];  // [H]
  const float* b_hh   = (const float*)d_in[5];  // [H]
  const float* W_fc   = (const float*)d_in[6];  // [O,H]
  const float* b_fc   = (const float*)d_in[7];  // [O]

  float* out = (float*)d_out;                   // [T,O] then [1,B,H]
  float* hT  = out + (size_t)Tt * Oo;

  // workspace carve (~340 MB; xt dominates)
  char* p = (char*)d_ws;
  float*  xt     = (float*)p;  p += (size_t)Tt * Bsz * Hh * sizeof(float);
  __bf16* xb     = (__bf16*)p; p += (size_t)BT * Dd * sizeof(__bf16);
  __bf16* wih_b  = (__bf16*)p; p += (size_t)Hh * Dd * sizeof(__bf16);
  __bf16* wfc_b  = (__bf16*)p; p += (size_t)Oo * Hh * sizeof(__bf16);
  __bf16* hbuf0  = (__bf16*)p; p += (size_t)Bsz * Hh * sizeof(__bf16);
  __bf16* hbuf1  = (__bf16*)p; p += (size_t)Bsz * Hh * sizeof(__bf16);
  __bf16* seq63  = (__bf16*)p; p += (size_t)Tt * Hh * sizeof(__bf16);
  unsigned* bar  = (unsigned*)p;

  // one-pass conversions (x fits L2 as bf16: 67 MB < 192 MB)
  cvt_f32_bf16<<<(BT * Dd / 8 + 255) / 256, 256, 0, stream>>>(x, xb, BT * Dd);
  cvt_f32_bf16<<<(Hh * Dd / 8 + 255) / 256, 256, 0, stream>>>(W_ih, wih_b,
                                                              Hh * Dd);
  cvt_f32_bf16<<<(Oo * Hh / 8 + 255) / 256, 256, 0, stream>>>(W_fc, wfc_b,
                                                              Oo * Hh);
  rnn_init<<<(Bsz * Hh + 255) / 256, 256, 0, stream>>>(hidden, hbuf0, bar);

  rnn_proj_wmma<<<dim3(Hh / 128, BT / 128), 256, 0, stream>>>(xb, wih_b, b_ih,
                                                              xt);
  rnn_scan_wmma<<<NWG, 256, 0, stream>>>(xt, W_hh, b_hh, hbuf0, hbuf1, seq63,
                                         hT, bar);
  rnn_fc_wmma<<<dim3(Oo / 128, Tt / 128), 256, 0, stream>>>(seq63, wfc_b, b_fc,
                                                            out);
}
